// CapsNet_2594160247458
// MI455X (gfx1250) — matrix operand order repaired and checked by reading, not compiled, not run on hardware
//
#include <hip/hip_runtime.h>
#include <hip/hip_bf16.h>

// ---------------------------------------------------------------------------
// CapsNet forward for MI455X (gfx1250, wave32).
// Dominant cost: PrimaryCaps conv (~98 GFLOP) -> implicit GEMM with
// v_wmma_f32_16x16x32_bf16. Activations stored NHWC bf16 so im2col K-chunks
// (ordered ky,kx,ci) are contiguous 32B loads.
// ---------------------------------------------------------------------------

typedef __attribute__((ext_vector_type(16))) __bf16        v16bf;
typedef __attribute__((ext_vector_type(8)))  float         v8f;
typedef __attribute__((ext_vector_type(4)))  unsigned int  v4u;

union FragBF { v4u q[2]; v16bf v; };   // 32 bytes: one 16x32/32x16 bf16 fragment per lane

#define PCK 20736          // K = 81*256 for primary-caps GEMM
#define EPS_SQ 1e-12f

__device__ __forceinline__ unsigned short f32_to_bf16(float f) {
    unsigned int u = __float_as_uint(f);
    unsigned int r = u + 0x7FFFu + ((u >> 16) & 1u);   // round-to-nearest-even
    return (unsigned short)(r >> 16);
}

// ---------------------------------------------------------------------------
// Conv1: [256,1,28,28] f32 -> NHWC bf16 [256,20,20,256], ReLU.
// One block per (b,y,x); 81-tap input window cached in LDS; thread = out chan.
// ---------------------------------------------------------------------------
__global__ __launch_bounds__(256) void conv1_k(const float* __restrict__ data,
                                               const float* __restrict__ cw,
                                               const float* __restrict__ cb,
                                               unsigned short* __restrict__ hN) {
    int sp = blockIdx.x;                 // b*400 + y*20 + x
    int b = sp / 400, yx = sp % 400, y = yx / 20, x = yx % 20;
    int co = threadIdx.x;
    __shared__ float win[81];
    if (co < 81) {
        int ky = co / 9, kx = co % 9;
        win[co] = data[b * 784 + (y + ky) * 28 + (x + kx)];
    }
    __syncthreads();
    float acc = cb[co];
    const float* wp = cw + co * 81;
#pragma unroll
    for (int k = 0; k < 81; ++k) acc = fmaf(win[k], wp[k], acc);
    acc = fmaxf(acc, 0.0f);
    hN[(size_t)sp * 256 + co] = f32_to_bf16(acc);
}

// ---------------------------------------------------------------------------
// Repack prim_w [co,ci,ky,kx] f32 -> bf16 A-matrix [co][k], k=(ky*9+kx)*256+ci.
// ---------------------------------------------------------------------------
__global__ void pack_w_k(const float* __restrict__ w, unsigned short* __restrict__ w2) {
    int i = blockIdx.x * blockDim.x + threadIdx.x;
    if (i >= 256 * PCK) return;
    int co = i / PCK, k = i % PCK;
    int pos = k >> 8, ci = k & 255;
    int ky = pos / 9, kx = pos - ky * 9;
    w2[i] = f32_to_bf16(w[((co * 256 + ci) * 81) + ky * 9 + kx]);
}

// ---------------------------------------------------------------------------
// PrimaryCaps implicit GEMM, bf16 WMMA. C[m=co][n=b*36+oy*6+ox], K=20736.
// Wave tile 64x32 (4x2 fragments, 8 WMMAs/K-step); 4 waves/WG -> 128x64 tile.
// ---------------------------------------------------------------------------
__global__ __launch_bounds__(128) void primary_gemm_wmma(
    const unsigned short* __restrict__ w2,    // [256][20736] bf16
    const unsigned short* __restrict__ hN,    // [256*400][256] bf16 (NHWC)
    const float* __restrict__ prim_b,
    float* __restrict__ u) {                  // [256][256][36] f32 (b-major)
    const int lane = threadIdx.x & 31;
    const int wave = threadIdx.x >> 5;
    const int wm = wave & 1, wn = wave >> 1;
    const int Mw = blockIdx.x * 128 + wm * 64;
    const int Nw = blockIdx.y * 64 + wn * 32;
    const int lr   = lane & 15;
    const int hiA  = (lane >> 4) * 16;    // A K-chunk base offset (0 | 16)
    const int hiB  = (lane >> 4) * 16;    // B K-chunk base offset (0 | 16)

    // Pre-decode B columns for this lane (n -> batch, oy, ox)
    int bI[2], oy[2], ox[2];
#pragma unroll
    for (int nt = 0; nt < 2; ++nt) {
        int n = Nw + nt * 16 + lr;
        bI[nt] = n / 36;
        int spc = n % 36;
        oy[nt] = spc / 6;
        ox[nt] = spc % 6;
    }
    const unsigned short* arow[4];
#pragma unroll
    for (int mt = 0; mt < 4; ++mt)
        arow[mt] = w2 + (size_t)(Mw + mt * 16 + lr) * PCK;

    v8f zacc = {0.f, 0.f, 0.f, 0.f, 0.f, 0.f, 0.f, 0.f};
    v8f acc[4][2];
#pragma unroll
    for (int mt = 0; mt < 4; ++mt)
#pragma unroll
        for (int nt = 0; nt < 2; ++nt) acc[mt][nt] = zacc;

    for (int k0 = 0; k0 < PCK; k0 += 32) {
        // B fragments (im2col on the fly): 16 contiguous ci at fixed (ky,kx)
        int kb  = k0 + hiB;
        int pos = kb >> 8, ci0 = kb & 255;
        int ky = pos / 9, kx = pos - ky * 9;
        FragBF bfr[2];
#pragma unroll
        for (int nt = 0; nt < 2; ++nt) {
            int iy = oy[nt] * 2 + ky, ix = ox[nt] * 2 + kx;
            const v4u* p = (const v4u*)(hN + (((bI[nt] * 20 + iy) * 20 + ix) << 8) + ci0);
            bfr[nt].q[0] = p[0];
            bfr[nt].q[1] = p[1];
        }
        // A fragments: two 8-element chunks of contiguous K
        FragBF afr[4];
#pragma unroll
        for (int mt = 0; mt < 4; ++mt) {
            const unsigned short* ap = arow[mt] + k0 + hiA;
            afr[mt].q[0] = *(const v4u*)(ap);
            afr[mt].q[1] = *(const v4u*)(ap + 8);
        }
#pragma unroll
        for (int mt = 0; mt < 4; ++mt)
#pragma unroll
            for (int nt = 0; nt < 2; ++nt)
                acc[mt][nt] = __builtin_amdgcn_wmma_f32_16x16x32_bf16(
                    false, afr[mt].v, false, bfr[nt].v,
                    (short)0, acc[mt][nt], false, false);
    }

    // Write-out: C layout -> lane<16: M=p, lane>=16: M=p+8; N = lane&15
#pragma unroll
    for (int mt = 0; mt < 4; ++mt) {
        int mbase = Mw + mt * 16 + ((lane >> 4) * 8);
#pragma unroll
        for (int nt = 0; nt < 2; ++nt) {
            int n = Nw + nt * 16 + lr;
            int b = n / 36, spc = n % 36;
#pragma unroll
            for (int p = 0; p < 8; ++p) {
                int co = mbase + p;
                u[(size_t)b * 9216 + co * 36 + spc] = acc[mt][nt][p] + prim_b[co];
            }
        }
    }
}

// ---------------------------------------------------------------------------
// Capsule squash over the 8-dim capsule axis (in place).
// ---------------------------------------------------------------------------
__global__ void squash_caps_k(float* __restrict__ u) {
    int cap = blockIdx.x * blockDim.x + threadIdx.x;
    if (cap >= 256 * 1152) return;
    float* p = u + (size_t)cap * 8;
    float t[8], sq = 0.f;
#pragma unroll
    for (int i = 0; i < 8; ++i) { t[i] = p[i]; sq += t[i] * t[i]; }
    float sc = sq / ((1.f + sq) * sqrtf(sq + EPS_SQ));
#pragma unroll
    for (int i = 0; i < 8; ++i) p[i] = t[i] * sc;
}

__global__ void zero_f32_k(float* p, int n) {
    int i = blockIdx.x * blockDim.x + threadIdx.x;
    if (i < n) p[i] = 0.f;
}

// softmax of b_ij over routes (axis 0); one block per class j.
__global__ __launch_bounds__(256) void softmax_routes_k(const float* __restrict__ b_ij,
                                                        float* __restrict__ c) {
    int j = blockIdx.x, t = threadIdx.x;
    __shared__ float red[256];
    float m = -1e30f;
    for (int r = t; r < 1152; r += 256) m = fmaxf(m, b_ij[r * 15 + j]);
    red[t] = m; __syncthreads();
    for (int s = 128; s > 0; s >>= 1) { if (t < s) red[t] = fmaxf(red[t], red[t + s]); __syncthreads(); }
    m = red[0]; __syncthreads();
    float sum = 0.f;
    for (int r = t; r < 1152; r += 256) sum += expf(b_ij[r * 15 + j] - m);
    red[t] = sum; __syncthreads();
    for (int s = 128; s > 0; s >>= 1) { if (t < s) red[t] += red[t + s]; __syncthreads(); }
    float Z = red[0];
    for (int r = t; r < 1152; r += 256) c[r * 15 + j] = expf(b_ij[r * 15 + j] - m) / Z;
}

// s[b,j,:] = sum_r c[r,j] * (W[r,j,:,:] @ u[b,r,:])  (u_hat recomputed on the fly)
__global__ __launch_bounds__(128) void route_s_k(const float* __restrict__ c,
                                                 const float* __restrict__ Wd,
                                                 const float* __restrict__ u,
                                                 float* __restrict__ s) {
    int bj = blockIdx.x, b = bj / 15, j = bj % 15;
    int t = threadIdx.x, o = t & 15, rc = t >> 4;
    __shared__ float red[128];
    float acc = 0.f;
    for (int r = rc; r < 1152; r += 8) {
        const float* wp = Wd + ((size_t)(r * 15 + j) * 16 + o) * 8;
        const float* up = u + (size_t)b * 9216 + r * 8;
        float d = 0.f;
#pragma unroll
        for (int i = 0; i < 8; ++i) d = fmaf(wp[i], up[i], d);
        acc = fmaf(c[r * 15 + j], d, acc);
    }
    red[t] = acc; __syncthreads();
    if (t < 16) {
        float sum = 0.f;
#pragma unroll
        for (int g = 0; g < 8; ++g) sum += red[g * 16 + t];
        s[(size_t)bj * 16 + t] = sum;
    }
}

// elementwise squash (reference squashes over trailing size-1 dim)
__global__ void route_v_k(const float* __restrict__ s, float* __restrict__ v, int n) {
    int i = blockIdx.x * blockDim.x + threadIdx.x;
    if (i >= n) return;
    float x = s[i], sq = x * x;
    v[i] = sq * x / ((1.f + sq) * sqrtf(sq + EPS_SQ));
}

// b_ij[r,j] += mean_b( sum_o u_hat[b,r,j,o] * v[b,j,o] ), u_hat recomputed
__global__ __launch_bounds__(256) void route_agree_k(const float* __restrict__ Wd,
                                                     const float* __restrict__ u,
                                                     const float* __restrict__ v,
                                                     float* __restrict__ b_ij) {
    int rj = blockIdx.x, r = rj / 15, j = rj % 15;
    int t = threadIdx.x;                       // t == batch index
    __shared__ float Wl[128];
    __shared__ float red[256];
    if (t < 128) Wl[t] = Wd[(size_t)rj * 128 + t];
    __syncthreads();
    const float* up = u + (size_t)t * 9216 + r * 8;
    const float* vp = v + ((size_t)t * 15 + j) * 16;
    float ul[8];
#pragma unroll
    for (int i = 0; i < 8; ++i) ul[i] = up[i];
    float dot = 0.f;
#pragma unroll
    for (int o = 0; o < 16; ++o) {
        float inner = 0.f;
#pragma unroll
        for (int i = 0; i < 8; ++i) inner = fmaf(Wl[o * 8 + i], ul[i], inner);
        dot = fmaf(vp[o], inner, dot);
    }
    red[t] = dot; __syncthreads();
    for (int s = 128; s > 0; s >>= 1) { if (t < s) red[t] += red[t + s]; __syncthreads(); }
    if (t == 0) b_ij[rj] += red[0] * (1.f / 256.f);
}

__global__ void classes_k(const float* __restrict__ v, float* __restrict__ cls) {
    int i = blockIdx.x * blockDim.x + threadIdx.x;
    if (i >= 3840) return;
    const float* vp = v + (size_t)i * 16;
    float sq = 0.f;
#pragma unroll
    for (int o = 0; o < 16; ++o) sq += vp[o] * vp[o];
    cls[i] = sqrtf(sq);
}

// column-wise (over batch) max and exp-sum for legacy softmax over axis 0
__global__ __launch_bounds__(256) void colstats_k(const float* __restrict__ cls,
                                                  float* __restrict__ colm,
                                                  float* __restrict__ colz) {
    int j = blockIdx.x, t = threadIdx.x;
    __shared__ float red[256];
    float val = cls[t * 15 + j];
    red[t] = val; __syncthreads();
    for (int s = 128; s > 0; s >>= 1) { if (t < s) red[t] = fmaxf(red[t], red[t + s]); __syncthreads(); }
    float m = red[0]; __syncthreads();
    red[t] = expf(val - m); __syncthreads();
    for (int s = 128; s > 0; s >>= 1) { if (t < s) red[t] += red[t + s]; __syncthreads(); }
    if (t == 0) { colm[j] = m; colz[j] = red[0]; }
}

// per-batch argmax over j of sm[b,j]=exp(cls-colm)/colz, then mask + rec_in
__global__ __launch_bounds__(256) void argmax_mask_k(const float* __restrict__ cls,
                                                     const float* __restrict__ colm,
                                                     const float* __restrict__ colz,
                                                     const float* __restrict__ v,
                                                     float* __restrict__ masked,
                                                     float* __restrict__ rec_in) {
    int b = threadIdx.x;
    float best = -1e30f; int bi = 0;
    for (int j = 0; j < 15; ++j) {
        float sm = expf(cls[b * 15 + j] - colm[j]) / colz[j];
        if (sm > best) { best = sm; bi = j; }     // first-max, matches jnp.argmax
    }
    for (int j = 0; j < 15; ++j) {
        float mk = (j == bi) ? 1.f : 0.f;
        masked[b * 15 + j] = mk;
        for (int o = 0; o < 16; ++o)
            rec_in[b * 240 + j * 16 + o] = v[((size_t)b * 15 + j) * 16 + o] * mk;
    }
}

// dense layer: out[b,n] = act(in[b,:] . W[n,:] + bias[n]); ACT 0=relu 1=sigmoid
template <int ACT>
__global__ void dense_k(const float* __restrict__ in, const float* __restrict__ W,
                        const float* __restrict__ bias, float* __restrict__ out,
                        int N, int K) {
    int idx = blockIdx.x * blockDim.x + threadIdx.x;
    if (idx >= 256 * N) return;
    int b = idx / N, n = idx % N;
    const float* ip = in + (size_t)b * K;
    const float* wp = W + (size_t)n * K;
    float acc = bias[n];
    for (int k = 0; k < K; ++k) acc = fmaf(ip[k], wp[k], acc);
    if (ACT == 0) acc = fmaxf(acc, 0.f);
    else          acc = 1.f / (1.f + expf(-acc));
    out[idx] = acc;
}

// ---------------------------------------------------------------------------
extern "C" void kernel_launch(void* const* d_in, const int* in_sizes, int n_in,
                              void* d_out, int out_size, void* d_ws, size_t ws_size,
                              hipStream_t stream) {
    const float* data   = (const float*)d_in[0];
    const float* conv_w = (const float*)d_in[1];
    const float* conv_b = (const float*)d_in[2];
    const float* prim_w = (const float*)d_in[3];
    const float* prim_b = (const float*)d_in[4];
    const float* Wd     = (const float*)d_in[5];
    const float* dw1    = (const float*)d_in[6];
    const float* db1    = (const float*)d_in[7];
    const float* dw2    = (const float*)d_in[8];
    const float* db2    = (const float*)d_in[9];
    const float* dw3    = (const float*)d_in[10];
    const float* db3    = (const float*)d_in[11];

    // workspace carve-up (256B aligned)
    char* w = (char*)d_ws;
    auto carve = [&](size_t bytes) { char* p = w; w += (bytes + 255) & ~(size_t)255; return p; };
    unsigned short* hN   = (unsigned short*)carve((size_t)256 * 400 * 256 * 2); // 52.4 MB NHWC bf16
    unsigned short* w2   = (unsigned short*)carve((size_t)256 * PCK * 2);       // 10.6 MB bf16 A
    float* u      = (float*)carve((size_t)256 * 9216 * 4);                      // primary caps (squashed in place)
    float* b_ij   = (float*)carve(17280 * 4);
    float* c      = (float*)carve(17280 * 4);
    float* s      = (float*)carve(61440 * 4);
    float* cls    = (float*)carve(3840 * 4);
    float* colm   = (float*)carve(64 * 4);
    float* colz   = (float*)carve(64 * 4);
    float* rec_in = (float*)carve(61440 * 4);
    float* h1     = (float*)carve((size_t)256 * 512 * 4);
    float* h2     = (float*)carve((size_t)256 * 1024 * 4);

    float* vout   = (float*)d_out;        // [256,15,16]
    float* recon  = vout + 61440;         // [256,784]
    float* masked = recon + 200704;       // [256,15]

    conv1_k<<<102400, 256, 0, stream>>>(data, conv_w, conv_b, hN);
    pack_w_k<<<(256 * PCK + 255) / 256, 256, 0, stream>>>(prim_w, w2);
    primary_gemm_wmma<<<dim3(2, 144), 128, 0, stream>>>(w2, hN, prim_b, u);
    squash_caps_k<<<(256 * 1152 + 255) / 256, 256, 0, stream>>>(u);
    zero_f32_k<<<(17280 + 255) / 256, 256, 0, stream>>>(b_ij, 17280);

    for (int it = 0; it < 3; ++it) {
        softmax_routes_k<<<15, 256, 0, stream>>>(b_ij, c);
        route_s_k<<<3840, 128, 0, stream>>>(c, Wd, u, s);
        route_v_k<<<(61440 + 255) / 256, 256, 0, stream>>>(s, vout, 61440);
        if (it < 2)
            route_agree_k<<<17280, 256, 0, stream>>>(Wd, u, vout, b_ij);
    }

    classes_k<<<(3840 + 255) / 256, 256, 0, stream>>>(vout, cls);
    colstats_k<<<15, 256, 0, stream>>>(cls, colm, colz);
    argmax_mask_k<<<1, 256, 0, stream>>>(cls, colm, colz, vout, masked, rec_in);

    dense_k<0><<<(256 * 512 + 255) / 256, 256, 0, stream>>>(rec_in, dw1, db1, h1, 512, 240);
    dense_k<0><<<(256 * 1024 + 255) / 256, 256, 0, stream>>>(h1, dw2, db2, h2, 1024, 512);
    dense_k<1><<<(256 * 784 + 255) / 256, 256, 0, stream>>>(h2, dw3, db3, recon, 784, 1024);
}
